// BiformerLayer_54030688583932
// MI455X (gfx1250) — compile-verified, matrix operand order checked by reading, not run on hardware
//
#include <hip/hip_runtime.h>
#include <hip/hip_bf16.h>
#include <math.h>

typedef __attribute__((ext_vector_type(16))) __bf16 v16bf;
typedef __attribute__((ext_vector_type(8)))  float  v8f;
typedef __attribute__((ext_vector_type(4)))  unsigned int u32x4;
typedef __attribute__((ext_vector_type(4)))  int v4i;

#define NB    8
#define CH    256
#define HH    56
#define WW_   56
#define HW    3136
#define TOK   25088      // NB*HW
#define NWIN  7
#define REG   49
#define HEADS 8
#define CD    32
#define TOPK  4
#define KV    256        // TOPK * 64
#define LDK   40         // padded 32-k LDS row (bank-conflict-free b128 frags)
#define LDP   264        // padded 256-k LDS row
#define LDS_  257        // padded softmax row (floats)

// ---------------------------------------------------------------------------
// CDNA5 async global->LDS staging (ASYNCcnt path) with safe fallback
// ---------------------------------------------------------------------------
#if defined(__has_builtin)
#if __has_builtin(__builtin_amdgcn_global_load_async_to_lds_b128) && \
    __has_builtin(__builtin_amdgcn_s_wait_asynccnt)
#define USE_ASYNC_LDS 1
#endif
#endif

__device__ inline void cp16_g2l(const void* g, void* l) {
#ifdef USE_ASYNC_LDS
  typedef __attribute__((address_space(1))) v4i GV4;
  typedef __attribute__((address_space(3))) v4i LV4;
  __builtin_amdgcn_global_load_async_to_lds_b128(
      (GV4*)(uintptr_t)g, (LV4*)l, 0, 0);
#else
  *(u32x4*)l = *(const u32x4*)g;
#endif
}
__device__ inline void async_fence() {
#ifdef USE_ASYNC_LDS
  __builtin_amdgcn_s_wait_asynccnt(0);
#endif
}

// ---------------------------------------------------------------------------
// CDNA5 16-bit fragment load (wave32): lane's 16 bf16 are two contiguous
// 16B runs at row + h*8 and row + 16 + h*8  ->  two ds_load_b128.
// Works for A (m,k)=base[m*ld+k] and for B given as B^T (n,k)=base[n*ld+k].
// ---------------------------------------------------------------------------
__device__ inline v16bf frag_ld(const __bf16* base, int ld) {
  int lane = threadIdx.x & 31;
  const __bf16* row = base + (lane & 15) * ld + ((lane >> 4) << 3);
  union { v16bf v; u32x4 q[2]; } u;
  u.q[0] = *(const u32x4*)(row);
  u.q[1] = *(const u32x4*)(row + 16);
  return u.v;
}

// ---------------------------------------------------------------------------
// bf16 WMMA GEMM: A[M][K] x BT[N][K] -> 128x128 tile, fused epilogues
// mode 0: outF = acc + bias
// mode 1: outH = bf16(gelu(acc + bias))
// mode 2: outF = resid + acc + bias          (resid may alias outF)
// mode 3: outH = bf16(y*gate + y*sigmoid(acc+bias)), y = resid  (SCSE)
// ---------------------------------------------------------------------------
__global__ __launch_bounds__(256) void k_gemm(
    const __bf16* __restrict__ A, const __bf16* __restrict__ BT,
    const float* __restrict__ bias, float* __restrict__ outF,
    __bf16* __restrict__ outH, const float* __restrict__ resid,
    const float* __restrict__ gate, int M, int K, int Ncol, int mode)
{
  __shared__ __bf16 sA[128 * LDK];
  __shared__ __bf16 sB[128 * LDK];
  const int tid = threadIdx.x;
  const int rowBase = blockIdx.y * 128;
  const int colBase = blockIdx.x * 128;
  const int lane = tid & 31, w = tid >> 5;
  const int wm = w & 3, wn = w >> 2;
  const int m0 = wm * 32, n0 = wn * 64;

  v8f acc[2][4];
#pragma unroll
  for (int i = 0; i < 2; ++i)
#pragma unroll
    for (int j = 0; j < 4; ++j) acc[i][j] = {};

  for (int k0 = 0; k0 < K; k0 += 32) {
#pragma unroll
    for (int i = 0; i < 2; ++i) {
      int chunk = i * 256 + tid;                 // 512 chunks of 8 bf16
      int r = chunk >> 2, c8 = (chunk & 3) * 8;  // tiles: 128 x 32
      cp16_g2l(&A[(size_t)(rowBase + r) * K + k0 + c8],  &sA[r * LDK + c8]);
      cp16_g2l(&BT[(size_t)(colBase + r) * K + k0 + c8], &sB[r * LDK + c8]);
    }
    if (k0 + 32 < K) {
      __builtin_prefetch(&A[(size_t)(rowBase + (tid >> 1)) * K + k0 + 32], 0, 0);
      __builtin_prefetch(&BT[(size_t)(colBase + (tid >> 1)) * K + k0 + 32], 0, 0);
    }
    async_fence();
    __syncthreads();
    v16bf a0 = frag_ld(sA + (m0 + 0) * LDK, LDK);
    v16bf a1 = frag_ld(sA + (m0 + 16) * LDK, LDK);
#pragma unroll
    for (int j = 0; j < 4; ++j) {
      v16bf b = frag_ld(sB + (n0 + j * 16) * LDK, LDK);
      acc[0][j] = __builtin_amdgcn_wmma_f32_16x16x32_bf16(false, a0, false, b, (short)0, acc[0][j], false, false);
      acc[1][j] = __builtin_amdgcn_wmma_f32_16x16x32_bf16(false, a1, false, b, (short)0, acc[1][j], false, false);
    }
    __syncthreads();
  }

  const int hl = lane >> 4, nn = lane & 15;
#pragma unroll
  for (int i = 0; i < 2; ++i)
#pragma unroll
    for (int j = 0; j < 4; ++j)
#pragma unroll
      for (int r = 0; r < 8; ++r) {
        int row = rowBase + m0 + i * 16 + r + 8 * hl;
        int col = colBase + n0 + j * 16 + nn;
        float v = acc[i][j][r] + bias[col];
        size_t off = (size_t)row * Ncol + col;
        if (mode == 0) {
          outF[off] = v;
        } else if (mode == 1) {
          outH[off] = (__bf16)(0.5f * v * (1.0f + erff(v * 0.70710678f)));
        } else if (mode == 2) {
          outF[off] = resid[off] + v;
        } else {
          float y = resid[off];
          float s = 1.0f / (1.0f + expf(-v));
          float g = gate[(row / HW) * CH + col];
          outH[off] = (__bf16)(y * g + y * s);
        }
      }
}

// ---------------------------------------------------------------------------
// Implicit-GEMM 3x3 conv (pad 1): A = gated NHWC bf16,
// BT = conv weights as [o][tap*256+c] bf16, out = NCHW fp32.
// ---------------------------------------------------------------------------
__global__ __launch_bounds__(256) void k_conv3(
    const __bf16* __restrict__ A, const __bf16* __restrict__ BT,
    const float* __restrict__ bias, float* __restrict__ out)
{
  __shared__ __bf16 sA[128 * LDK];
  __shared__ __bf16 sB[128 * LDK];
  const int tid = threadIdx.x;
  const int rowBase = blockIdx.y * 128;
  const int colBase = blockIdx.x * 128;
  const int lane = tid & 31, w = tid >> 5;
  const int wm = w & 3, wn = w >> 2;
  const int m0 = wm * 32, n0 = wn * 64;

  v8f acc[2][4];
#pragma unroll
  for (int i = 0; i < 2; ++i)
#pragma unroll
    for (int j = 0; j < 4; ++j) acc[i][j] = {};

  for (int s = 0; s < 72; ++s) {   // 9 taps x (256/32) channel slabs
    int tap = s >> 3, c0 = (s & 7) * 32;
    int kh = tap / 3 - 1, kw = tap % 3 - 1;
#pragma unroll
    for (int i = 0; i < 2; ++i) {
      int chunk = i * 256 + tid;
      int r = chunk >> 2, c8 = (chunk & 3) * 8;
      int token = rowBase + r;
      int n = token / HW, rem = token % HW;
      int h = rem / WW_ + kh, wx = rem % WW_ + kw;
      if ((unsigned)h < (unsigned)HH && (unsigned)wx < (unsigned)WW_) {
        cp16_g2l(&A[((size_t)(n * HW + h * WW_ + wx)) * CH + c0 + c8],
                 &sA[r * LDK + c8]);
      } else {
        u32x4 z = {0u, 0u, 0u, 0u};
        *(u32x4*)&sA[r * LDK + c8] = z;
      }
      cp16_g2l(&BT[(size_t)(colBase + r) * 2304 + tap * CH + c0 + c8],
               &sB[r * LDK + c8]);
    }
    async_fence();
    __syncthreads();
    v16bf a0 = frag_ld(sA + (m0 + 0) * LDK, LDK);
    v16bf a1 = frag_ld(sA + (m0 + 16) * LDK, LDK);
#pragma unroll
    for (int j = 0; j < 4; ++j) {
      v16bf b = frag_ld(sB + (n0 + j * 16) * LDK, LDK);
      acc[0][j] = __builtin_amdgcn_wmma_f32_16x16x32_bf16(false, a0, false, b, (short)0, acc[0][j], false, false);
      acc[1][j] = __builtin_amdgcn_wmma_f32_16x16x32_bf16(false, a1, false, b, (short)0, acc[1][j], false, false);
    }
    __syncthreads();
  }

  const int hl = lane >> 4, nn = lane & 15;
#pragma unroll
  for (int i = 0; i < 2; ++i)
#pragma unroll
    for (int j = 0; j < 4; ++j)
#pragma unroll
      for (int r = 0; r < 8; ++r) {
        int row = rowBase + m0 + i * 16 + r + 8 * hl;
        int col = colBase + n0 + j * 16 + nn;
        int n = row / HW, rem = row % HW;
        out[((size_t)(n * CH + col)) * HW + rem] = acc[i][j][r] + bias[col];
      }
}

// ---------------------------------------------------------------------------
// Routed attention: one block (128 thr / 4 waves) per (head, region, batch)
// ---------------------------------------------------------------------------
__global__ __launch_bounds__(128) void k_attn(
    const float* __restrict__ qkv, const int* __restrict__ idx,
    float* __restrict__ out)
{
  __shared__ __bf16 sQ[64 * LDK];    //  5 KB
  __shared__ __bf16 sK[KV * LDK];    // 20 KB  (kv-major, chan contiguous)
  __shared__ __bf16 sVT[CD * LDP];   // 16.5 KB (chan-major, kv contiguous)
  __shared__ float  sS[64 * LDS_];   // 64 KB
  __shared__ __bf16 sP[64 * LDP];    // 33 KB
  __shared__ float  red[256];
  const int head = blockIdx.x, p = blockIdx.y, n = blockIdx.z;
  const int tid = threadIdx.x;
  const int jy = p / NWIN, jx = p % NWIN;
  const float scale = 0.0625f;       // C^-0.5

  for (int e = tid; e < 64 * 32; e += 128) {
    int qi = e >> 5, c = e & 31;
    int h = jy * 8 + (qi >> 3), wx = jx * 8 + (qi & 7);
    float v = qkv[((size_t)(n * HW + h * WW_ + wx)) * 768 + head * CD + c];
    sQ[qi * LDK + c] = (__bf16)(v * scale);
  }
  const int* id4 = idx + (n * REG + p) * TOPK;
  for (int e = tid; e < KV * 32; e += 128) {
    int row = e >> 5, c = e & 31;
    int t = row >> 6, pi = row & 63;
    int r = id4[t];
    int h = (r / NWIN) * 8 + (pi >> 3), wx = (r % NWIN) * 8 + (pi & 7);
    const float* base = qkv + ((size_t)(n * HW + h * WW_ + wx)) * 768 + head * CD + c;
    sK[row * LDK + c] = (__bf16)base[256];
    sVT[c * LDP + row] = (__bf16)base[512];
  }
  __syncthreads();

  const int lane = tid & 31, w = tid >> 5;
  const int hl = lane >> 4, nn = lane & 15;
  {
    v16bf a = frag_ld(sQ + w * 16 * LDK, LDK);   // wave's 16-row Q strip
#pragma unroll
    for (int ct = 0; ct < 16; ++ct) {
      v8f acc = {};
      v16bf b = frag_ld(sK + ct * 16 * LDK, LDK);
      acc = __builtin_amdgcn_wmma_f32_16x16x32_bf16(false, a, false, b, (short)0, acc, false, false);
#pragma unroll
      for (int r = 0; r < 8; ++r)
        sS[(w * 16 + r + 8 * hl) * LDS_ + ct * 16 + nn] = acc[r];
    }
  }
  __syncthreads();

  // softmax: 2 threads per row (row = tid>>1, halves of 128 cols)
  {
    int row = tid >> 1, half = tid & 1;
    float* rowp = sS + row * LDS_ + half * 128;
    float mx = -1e30f;
    for (int c = 0; c < 128; ++c) mx = fmaxf(mx, rowp[c]);
    red[tid] = mx; __syncthreads();
    mx = fmaxf(red[row * 2], red[row * 2 + 1]);
    float sum = 0.0f;
    for (int c = 0; c < 128; ++c) { float e = expf(rowp[c] - mx); rowp[c] = e; sum += e; }
    red[tid] = sum; __syncthreads();
    float inv = 1.0f / (red[row * 2] + red[row * 2 + 1]);
    __bf16* pp = sP + row * LDP + half * 128;
    for (int c = 0; c < 128; ++c) pp[c] = (__bf16)(rowp[c] * inv);
  }
  __syncthreads();

  {
    v8f acc2[2]; acc2[0] = {}; acc2[1] = {};
#pragma unroll
    for (int kk = 0; kk < 8; ++kk) {
      v16bf a = frag_ld(sP + w * 16 * LDP + kk * 32, LDP);
#pragma unroll
      for (int j = 0; j < 2; ++j) {
        v16bf b = frag_ld(sVT + j * 16 * LDP + kk * 32, LDP);
        acc2[j] = __builtin_amdgcn_wmma_f32_16x16x32_bf16(false, a, false, b, (short)0, acc2[j], false, false);
      }
    }
#pragma unroll
    for (int j = 0; j < 2; ++j)
#pragma unroll
      for (int r = 0; r < 8; ++r) {
        int qi = w * 16 + r + 8 * hl;
        int h = jy * 8 + (qi >> 3), wx = jx * 8 + (qi & 7);
        out[((size_t)(n * HW + h * WW_ + wx)) * CH + head * CD + j * 16 + nn] = acc2[j][r];
      }
  }
}

// ---------------------------------------------------------------------------
// Small VALU kernels
// ---------------------------------------------------------------------------
__global__ __launch_bounds__(256) void k_to_nhwc(const float* __restrict__ s,
                                                 float* __restrict__ d) {
  int i = blockIdx.x * 256 + threadIdx.x;          // NB*CH*HW
  int n = i / (CH * HW), rem = i % (CH * HW), c = rem / HW, hw = rem % HW;
  d[((size_t)(n * HW + hw)) * CH + c] = s[i];
}

__global__ __launch_bounds__(256) void k_layernorm(
    const float* __restrict__ x, const float* __restrict__ w,
    const float* __restrict__ b, __bf16* __restrict__ out) {
  __shared__ float red[256];
  int t = blockIdx.x, c = threadIdx.x;
  float v = x[(size_t)t * CH + c];
  red[c] = v; __syncthreads();
  for (int s = 128; s > 0; s >>= 1) { if (c < s) red[c] += red[c + s]; __syncthreads(); }
  float mu = red[0] * (1.0f / CH); __syncthreads();
  float dd = v - mu;
  red[c] = dd * dd; __syncthreads();
  for (int s = 128; s > 0; s >>= 1) { if (c < s) red[c] += red[c + s]; __syncthreads(); }
  float var = red[0] * (1.0f / CH);
  out[(size_t)t * CH + c] = (__bf16)(dd * rsqrtf(var + 1e-6f) * w[c] + b[c]);
}

__global__ __launch_bounds__(256) void k_region_stats(
    const float* __restrict__ qkv, float* __restrict__ qwin,
    float* __restrict__ kwin) {
  int p = blockIdx.x, n = blockIdx.y, c = threadIdx.x;
  int jy = p / NWIN, jx = p % NWIN;
  float sq = 0.f, sk = 0.f;
  for (int pi = 0; pi < 64; ++pi) {
    int h = jy * 8 + (pi >> 3), wx = jx * 8 + (pi & 7);
    const float* base = qkv + ((size_t)(n * HW + h * WW_ + wx)) * 768;
    sq += base[c]; sk += base[256 + c];
  }
  qwin[(size_t)(n * REG + p) * CH + c] = sq * (1.0f / 64.0f);
  kwin[(size_t)(n * REG + p) * CH + c] = sk * (1.0f / 64.0f);
}

__global__ __launch_bounds__(64) void k_topk(
    const float* __restrict__ qwin, const float* __restrict__ kwin,
    int* __restrict__ idxo) {
  __shared__ float lg[REG];
  int p = blockIdx.x, n = blockIdx.y, t = threadIdx.x;
  if (t < REG) {
    const float* q = qwin + (size_t)(n * REG + p) * CH;
    const float* k = kwin + (size_t)(n * REG + t) * CH;
    float s = 0.f;
    for (int c = 0; c < CH; ++c) s += q[c] * k[c];
    lg[t] = s * 0.0625f;
  }
  __syncthreads();
  if (t == 0) {
    for (int sel = 0; sel < TOPK; ++sel) {
      int best = 0; float bv = -1e30f;
      for (int j = 0; j < REG; ++j)
        if (lg[j] > bv) { bv = lg[j]; best = j; }
      idxo[(n * REG + p) * TOPK + sel] = best;
      lg[best] = -1e30f;
    }
  }
}

// LEPE 5x5 depthwise on V (qkv offset 512) + attention output -> bf16 A for WO
__global__ __launch_bounds__(256) void k_lepe_add(
    const float* __restrict__ qkv, const float* __restrict__ attn,
    const float* __restrict__ lw, const float* __restrict__ lb,
    __bf16* __restrict__ out) {
  int tkn = blockIdx.x, c = threadIdx.x;
  int n = tkn / HW, rem = tkn % HW, h = rem / WW_, wx = rem % WW_;
  const float* wk = lw + (size_t)c * 25;
  float acc = lb[c];
  for (int ky = 0; ky < 5; ++ky) {
    int hh2 = h + ky - 2;
    if ((unsigned)hh2 >= (unsigned)HH) continue;
    for (int kx = 0; kx < 5; ++kx) {
      int ww2 = wx + kx - 2;
      if ((unsigned)ww2 >= (unsigned)WW_) continue;
      acc += qkv[((size_t)(n * HW + hh2 * WW_ + ww2)) * 768 + 512 + c] * wk[ky * 5 + kx];
    }
  }
  out[(size_t)tkn * CH + c] = (__bf16)(attn[(size_t)tkn * CH + c] + acc);
}

__global__ __launch_bounds__(256) void k_pool(const float* __restrict__ xh,
                                              float* __restrict__ pooled) {
  int n = blockIdx.x, c = threadIdx.x;
  const float* p = xh + (size_t)n * HW * CH + c;
  float s = 0.f;
  for (int t = 0; t < HW; ++t) s += p[(size_t)t * CH];
  pooled[n * CH + c] = s * (1.0f / HW);
}

__global__ __launch_bounds__(256) void k_cse(
    const float* __restrict__ pooled, const float* __restrict__ w1,
    const float* __restrict__ b1, const float* __restrict__ w2,
    const float* __restrict__ b2, float* __restrict__ gate) {
  __shared__ float c1[16];
  int n = blockIdx.x, t = threadIdx.x;
  if (t < 16) {
    float s = b1[t];
    for (int c = 0; c < CH; ++c) s += pooled[n * CH + c] * w1[t * CH + c];
    c1[t] = fmaxf(s, 0.f);
  }
  __syncthreads();
  float s = b2[t];
  for (int j = 0; j < 16; ++j) s += c1[j] * w2[t * 16 + j];
  gate[n * CH + t] = 1.0f / (1.0f + expf(-s));
}

__global__ __launch_bounds__(256) void k_convert(const float* __restrict__ s,
                                                 __bf16* __restrict__ d, int nE) {
  int i = blockIdx.x * 256 + threadIdx.x;
  if (i < nE) d[i] = (__bf16)s[i];
}

// BT[n*K + k] = W[k*N + n]  (weights -> transposed bf16 for K-major B frags)
__global__ __launch_bounds__(256) void k_convert_T(const float* __restrict__ s,
                                                   __bf16* __restrict__ d,
                                                   int K, int N) {
  int i = blockIdx.x * 256 + threadIdx.x;
  if (i >= K * N) return;
  int n = i / K, k = i - n * K;
  d[i] = (__bf16)s[(size_t)k * N + n];
}

// conv weights [o][c][tap] -> BT[o][tap*256 + c]
__global__ __launch_bounds__(256) void k_conv_reorg(const float* __restrict__ s,
                                                    __bf16* __restrict__ d) {
  int i = blockIdx.x * 256 + threadIdx.x;  // 589824 = 256*2304
  if (i >= 256 * 2304) return;
  int o = i / 2304, rem = i % 2304, tap = rem >> 8, c = rem & 255;
  d[i] = (__bf16)s[(o * 256 + c) * 9 + tap];
}

// ---------------------------------------------------------------------------
extern "C" void kernel_launch(void* const* d_in, const int* in_sizes, int n_in,
                              void* d_out, int out_size, void* d_ws, size_t ws_size,
                              hipStream_t stream) {
  const float* x      = (const float*)d_in[0];
  const float* ln1_w  = (const float*)d_in[1];
  const float* ln1_b  = (const float*)d_in[2];
  const float* qkv_w  = (const float*)d_in[3];
  const float* qkv_b  = (const float*)d_in[4];
  const float* lepe_w = (const float*)d_in[5];
  const float* lepe_b = (const float*)d_in[6];
  const float* wo_w   = (const float*)d_in[7];
  const float* wo_b   = (const float*)d_in[8];
  const float* ln2_w  = (const float*)d_in[9];
  const float* ln2_b  = (const float*)d_in[10];
  const float* mlp1_w = (const float*)d_in[11];
  const float* mlp1_b = (const float*)d_in[12];
  const float* mlp2_w = (const float*)d_in[13];
  const float* mlp2_b = (const float*)d_in[14];
  const float* cse1_w = (const float*)d_in[15];
  const float* cse1_b = (const float*)d_in[16];
  const float* cse2_w = (const float*)d_in[17];
  const float* cse2_b = (const float*)d_in[18];
  const float* sse_w  = (const float*)d_in[19];
  const float* sse_b  = (const float*)d_in[20];
  const float* conv_w = (const float*)d_in[21];
  const float* conv_b = (const float*)d_in[22];

  char* ws = (char*)d_ws;
  size_t off = 0;
#define CARVE(T, NAME, BYTES) \
  T NAME = (T)(ws + off); off += (((size_t)(BYTES) + 255) & ~(size_t)255);

  CARVE(float*,   xh,     (size_t)TOK * CH * 4)       // NHWC activations
  CARVE(__bf16*,  xnb,    (size_t)TOK * CH * 2)       // LN output (bf16)
  CARVE(float*,   qkvb,   (size_t)TOK * 768 * 4)      // QKV fp32 (reused as MLP hidden bf16)
  CARVE(float*,   attn,   (size_t)TOK * CH * 4)
  CARVE(__bf16*,  woin,   (size_t)TOK * CH * 2)       // WO A-matrix (reused as gated y)
  CARVE(float*,   qwin,   (size_t)NB * REG * CH * 4)
  CARVE(float*,   kwin,   (size_t)NB * REG * CH * 4)
  CARVE(int*,     idxb,   (size_t)NB * REG * TOPK * 4)
  CARVE(float*,   pooled, (size_t)NB * CH * 4)
  CARVE(float*,   gate,   (size_t)NB * CH * 4)
  CARVE(__bf16*,  wb,     (size_t)256 * 2304 * 2)     // weight staging (bf16, BT)
#undef CARVE
  __bf16* hbf = (__bf16*)qkvb;   // MLP hidden aliases QKV buffer (51MB <= 77MB)
  __bf16* yg  = woin;            // gated activations alias WO input buffer

  // NCHW -> NHWC
  k_to_nhwc<<<TOK, 256, 0, stream>>>(x, xh);

  for (int i = 0; i < 4; ++i) {
    // LN1 -> bf16
    k_layernorm<<<TOK, 256, 0, stream>>>(xh, ln1_w + i * CH, ln1_b + i * CH, xnb);
    // QKV GEMM  (BT = qkv_w^T : [768][256])
    k_convert_T<<<768, 256, 0, stream>>>(qkv_w + (size_t)i * CH * 768, wb, CH, 768);
    k_gemm<<<dim3(6, 196), 256, 0, stream>>>(xnb, wb, qkv_b + i * 768,
                                             qkvb, nullptr, nullptr, nullptr,
                                             TOK, CH, 768, 0);
    // routing
    k_region_stats<<<dim3(REG, NB), 256, 0, stream>>>(qkvb, qwin, kwin);
    k_topk<<<dim3(REG, NB), 64, 0, stream>>>(qwin, kwin, idxb);
    // attention
    k_attn<<<dim3(HEADS, REG, NB), 128, 0, stream>>>(qkvb, idxb, attn);
    // LEPE + add -> bf16 WO input
    k_lepe_add<<<TOK, 256, 0, stream>>>(qkvb, attn, lepe_w + (size_t)i * CH * 25,
                                        lepe_b + i * CH, woin);
    // WO GEMM with residual into xh
    k_convert_T<<<256, 256, 0, stream>>>(wo_w + (size_t)i * CH * CH, wb, CH, CH);
    k_gemm<<<dim3(2, 196), 256, 0, stream>>>(woin, wb, wo_b + i * CH,
                                             xh, nullptr, xh, nullptr,
                                             TOK, CH, CH, 2);
    // LN2 -> bf16
    k_layernorm<<<TOK, 256, 0, stream>>>(xh, ln2_w + i * CH, ln2_b + i * CH, xnb);
    // MLP1 (GELU, bf16 out)
    k_convert_T<<<1024, 256, 0, stream>>>(mlp1_w + (size_t)i * CH * 1024, wb, CH, 1024);
    k_gemm<<<dim3(8, 196), 256, 0, stream>>>(xnb, wb, mlp1_b + i * 1024,
                                             nullptr, hbf, nullptr, nullptr,
                                             TOK, CH, 1024, 1);
    // MLP2 with residual into xh
    k_convert_T<<<1024, 256, 0, stream>>>(mlp2_w + (size_t)i * 1024 * CH, wb, 1024, CH);
    k_gemm<<<dim3(2, 196), 256, 0, stream>>>(hbf, wb, mlp2_b + i * CH,
                                             xh, nullptr, xh, nullptr,
                                             TOK, 1024, CH, 2);
  }

  // SCSE gating
  k_pool<<<NB, 256, 0, stream>>>(xh, pooled);
  k_cse<<<NB, 256, 0, stream>>>(pooled, cse1_w, cse1_b, cse2_w, cse2_b, gate);
  k_convert<<<TOK, 256, 0, stream>>>(xh, xnb, TOK * CH);  // y -> bf16 A
  k_convert<<<256, 256, 0, stream>>>(sse_w, wb, CH * CH); // sse_w is already BT [o][c]
  k_gemm<<<dim3(2, 196), 256, 0, stream>>>(xnb, wb, sse_b,
                                           nullptr, yg, xh, gate,
                                           TOK, CH, CH, 3);
  // Final 3x3 conv -> NCHW output
  k_conv_reorg<<<2304, 256, 0, stream>>>(conv_w, wb);
  k_conv3<<<dim3(2, 196), 256, 0, stream>>>(yg, wb, conv_b, (float*)d_out);
}
